// Mamba2Mixer_5360119185611
// MI455X (gfx1250) — matrix operand order repaired and checked by reading, not compiled, not run on hardware
//
#include <hip/hip_runtime.h>
#include <hip/hip_bf16.h>

// ---------------------------------------------------------------------------
// Mamba2 mixer for MI455X (gfx1250, wave32, WMMA).
// Pipeline: fp32->bf16 converts -> in_proj WMMA GEMM -> conv+SiLU -> softplus
//           -> sequential SSM scan (LDS-staged, register state) -> gated
//           RMSNorm (emit bf16) -> out_proj WMMA GEMM -> d_out (fp32).
// ---------------------------------------------------------------------------

#define HID_      2688
#define H_        64
#define P_        64
#define N_        128
#define G_        8
#define KC_       4
#define I_        (H_ * P_)              // 4096
#define GN_       (G_ * N_)              // 1024
#define CONV_DIM_ (I_ + 2 * GN_)         // 6144
#define TOTAL_    (2 * I_ + 2 * GN_ + H_) // 10304
#define B_        2
#define T_        2048
#define BT_       (B_ * T_)              // 4096
#define EPS_      1e-5f

typedef __bf16 bf16;
typedef __attribute__((ext_vector_type(8)))  __bf16 v8bf;
typedef __attribute__((ext_vector_type(16))) __bf16 v16bf;
typedef __attribute__((ext_vector_type(8)))  float  v8f;

__device__ __forceinline__ unsigned short f32_bf16_rne(float f) {
    union { float f; unsigned u; } v; v.f = f;
    unsigned u = v.u;
    unsigned r = u + 0x7FFFu + ((u >> 16) & 1u);   // round-to-nearest-even
    return (unsigned short)(r >> 16);
}

// ---------------------------------------------------------------------------
// fp32 -> bf16 conversion (grid-stride)
// ---------------------------------------------------------------------------
__global__ __launch_bounds__(256)
void f32_to_bf16_kernel(const float* __restrict__ src,
                        unsigned short* __restrict__ dst, size_t n) {
    size_t stride = (size_t)gridDim.x * blockDim.x;
    for (size_t i = (size_t)blockIdx.x * blockDim.x + threadIdx.x; i < n; i += stride)
        dst[i] = f32_bf16_rne(src[i]);
}

// ---------------------------------------------------------------------------
// bf16 WMMA GEMM: C[M,N] = A[M,K] * B[N,K]^T  (both K-major row-major)
// One wave computes a 64x64 tile as 4x4 grid of 16x16 WMMAs; 8 waves/block.
// ---------------------------------------------------------------------------
__global__ __launch_bounds__(256)
void gemm_bf16_wmma(const bf16* __restrict__ A, const bf16* __restrict__ Bw,
                    float* __restrict__ C,
                    int M, int Nn, int Kd, int lda, int ldb, int ldc) {
    const int lane = threadIdx.x & 31;
    const int wave = threadIdx.x >> 5;
    const int tiles_n = Nn >> 6;
    const int tiles_m = M >> 6;
    const int tile = blockIdx.x * 8 + wave;
    if (tile >= tiles_m * tiles_n) return;
    const int tm = tile / tiles_n;
    const int tn = tile - tm * tiles_n;
    const int row0 = tm << 6;
    const int col0 = tn << 6;

    v8f acc[4][4];
#pragma unroll
    for (int i = 0; i < 4; ++i)
#pragma unroll
        for (int j = 0; j < 4; ++j)
            acc[i][j] = (v8f){0.f, 0.f, 0.f, 0.f, 0.f, 0.f, 0.f, 0.f};

    // A 16x32 bf16 layout: lanes<16 hold K 0..7 & 16..23 of row M=lane&15;
    // lanes>=16 hold K 8..15 & 24..31.
    const int am  = row0 + (lane & 15);
    const int akb = (lane >> 4) << 3;     // 0 or 8
    // B 32x16 bf16 layout: column N=lane&15; lanes<16 hold K 0..7 & 16..23,
    // lanes>=16 hold K 8..15 & 24..31.
    const int bn  = col0 + (lane & 15);
    const int bkb = (lane >> 4) << 3;     // 0 or 8

    for (int k0 = 0; k0 < Kd; k0 += 32) {
        v16bf afrag[4], bfrag[4];
#pragma unroll
        for (int s = 0; s < 4; ++s) {
            const bf16* ap = A + (size_t)(am + s * 16) * lda + (k0 + akb);
            v8bf lo = *(const v8bf*)ap;
            v8bf hi = *(const v8bf*)(ap + 16);
            afrag[s] = __builtin_shufflevector(lo, hi,
                0, 1, 2, 3, 4, 5, 6, 7, 8, 9, 10, 11, 12, 13, 14, 15);
        }
#pragma unroll
        for (int s = 0; s < 4; ++s) {
            const bf16* bp = Bw + (size_t)(bn + s * 16) * ldb + (k0 + bkb);
            v8bf lo = *(const v8bf*)bp;
            v8bf hi = *(const v8bf*)(bp + 16);
            bfrag[s] = __builtin_shufflevector(lo, hi,
                0, 1, 2, 3, 4, 5, 6, 7, 8, 9, 10, 11, 12, 13, 14, 15);
        }
        if (k0 + 32 < Kd) {   // hint next K-slab -> global_prefetch_b8
            __builtin_prefetch((const void*)(A + (size_t)am * lda + (k0 + 32 + akb)), 0, 0);
            __builtin_prefetch((const void*)(Bw + (size_t)bn * ldb + (k0 + 32 + bkb)), 0, 0);
        }
#pragma unroll
        for (int i = 0; i < 4; ++i)
#pragma unroll
            for (int j = 0; j < 4; ++j)
                acc[i][j] = __builtin_amdgcn_wmma_f32_16x16x32_bf16(
                    false, afrag[i], false, bfrag[j],
                    (short)0, acc[i][j], false, false);
    }

    // C/D f32 layout: VGPR r -> row r (lanes<16) or r+8 (lanes>=16), col=lane&15
    const int crow = (lane < 16) ? 0 : 8;
    const int ccol = lane & 15;
#pragma unroll
    for (int i = 0; i < 4; ++i)
#pragma unroll
        for (int j = 0; j < 4; ++j) {
            float* cp = C + (size_t)(row0 + i * 16 + crow) * ldc
                          + (col0 + j * 16 + ccol);
#pragma unroll
            for (int r = 0; r < 8; ++r)
                cp[(size_t)r * ldc] = acc[i][j][r];
        }
}

// ---------------------------------------------------------------------------
// Depthwise causal conv (K=4) + SiLU over the xBC slice of proj
// ---------------------------------------------------------------------------
__global__ __launch_bounds__(256)
void conv_silu_kernel(const float* __restrict__ proj,
                      const float* __restrict__ conv_w,
                      const float* __restrict__ conv_b,
                      float* __restrict__ out) {
    size_t idx = (size_t)blockIdx.x * blockDim.x + threadIdx.x;
    if (idx >= (size_t)BT_ * CONV_DIM_) return;
    const int c  = (int)(idx % CONV_DIM_);
    const int bt = (int)(idx / CONV_DIM_);
    const int t  = bt % T_;
    const int b  = bt / T_;
    float acc = conv_b[c];
#pragma unroll
    for (int i = 0; i < KC_; ++i) {
        const int tt = t - (KC_ - 1) + i;
        if (tt >= 0)
            acc += proj[(size_t)(b * T_ + tt) * TOTAL_ + I_ + c] * conv_w[c * KC_ + i];
    }
    out[idx] = acc / (1.f + __expf(-acc));   // SiLU
}

// ---------------------------------------------------------------------------
// dt softplus: dt_sp = softplus(proj[..., I+CONV_DIM + h] + dt_bias[h])
// ---------------------------------------------------------------------------
__global__ __launch_bounds__(256)
void dt_softplus_kernel(const float* __restrict__ proj,
                        const float* __restrict__ dt_bias,
                        float* __restrict__ dtsp) {
    size_t idx = (size_t)blockIdx.x * blockDim.x + threadIdx.x;
    if (idx >= (size_t)BT_ * H_) return;
    const int h  = (int)(idx % H_);
    const int bt = (int)(idx / H_);
    const float z = proj[(size_t)bt * TOTAL_ + I_ + CONV_DIM_ + h] + dt_bias[h];
    dtsp[idx] = (z > 20.f) ? z : log1pf(__expf(z));
}

// ---------------------------------------------------------------------------
// Sequential SSM scan. One block per (b,h). 256 threads: thread = p*4 + nq,
// each holds 32 of the N=128 state elements in registers. B/C/x staged in LDS.
// ---------------------------------------------------------------------------
__global__ __launch_bounds__(256)
void ssm_scan_kernel(const float* __restrict__ conv,
                     const float* __restrict__ dtsp,
                     const float* __restrict__ Av,
                     const float* __restrict__ Dv,
                     float* __restrict__ yout) {
    __shared__ float xs[P_];
    __shared__ float Bs[N_];
    __shared__ float Cs[N_];

    const int b   = blockIdx.x / H_;
    const int h   = blockIdx.x % H_;
    const int g   = h / (H_ / G_);
    const int tid = threadIdx.x;
    const int p   = tid >> 2;
    const int nq  = tid & 3;
    const int nb  = nq * 32;
    const float Ah = Av[h];
    const float Dh = Dv[h];

    float hstate[32];
#pragma unroll
    for (int i = 0; i < 32; ++i) hstate[i] = 0.f;

    for (int t = 0; t < T_; ++t) {
        const int bt = b * T_ + t;
        const float* base = conv + (size_t)bt * CONV_DIM_;
        if (tid < 64) {
            xs[tid]      = base[h * P_ + tid];
            Cs[64 + tid] = base[I_ + GN_ + g * N_ + 64 + tid];
        } else if (tid < 192) {
            Bs[tid - 64] = base[I_ + g * N_ + (tid - 64)];
        } else {
            Cs[tid - 192] = base[I_ + GN_ + g * N_ + (tid - 192)];
        }
        __syncthreads();

        const float dtv   = dtsp[(size_t)bt * H_ + h];
        const float Ab    = __expf(Ah * dtv);
        const float xv    = xs[p];
        const float scale = dtv * xv;
        float y = 0.f;
#pragma unroll
        for (int i = 0; i < 32; ++i) {
            const float hv = Ab * hstate[i] + scale * Bs[nb + i];
            hstate[i] = hv;
            y += hv * Cs[nb + i];
        }
        // reduce across the 4 lanes covering N for this p (lane groups of 4)
        y += __shfl_xor(y, 1, 32);
        y += __shfl_xor(y, 2, 32);
        if (nq == 0)
            yout[(size_t)bt * I_ + h * P_ + p] = y + Dh * xv;
        __syncthreads();
    }
}

// ---------------------------------------------------------------------------
// Gated RMSNorm: y *= silu(gate); y *= rsqrt(mean(y^2)+eps)*w; emit bf16
// One block per (b,t) row of I=4096; 16 elements/thread.
// ---------------------------------------------------------------------------
__global__ __launch_bounds__(256)
void gate_norm_kernel(const float* __restrict__ yscan,
                      const float* __restrict__ proj,
                      const float* __restrict__ norm_w,
                      unsigned short* __restrict__ ynorm) {
    __shared__ float red[256];
    const int row = blockIdx.x;
    const int tid = threadIdx.x;
    float tv[16];
    float ss = 0.f;
#pragma unroll
    for (int j = 0; j < 16; ++j) {
        const int i = tid + j * 256;
        const float gv = proj[(size_t)row * TOTAL_ + i];
        const float yv = yscan[(size_t)row * I_ + i] * (gv / (1.f + __expf(-gv)));
        tv[j] = yv;
        ss += yv * yv;
    }
    red[tid] = ss;
    __syncthreads();
    for (int s = 128; s > 0; s >>= 1) {
        if (tid < s) red[tid] += red[tid + s];
        __syncthreads();
    }
    const float scale = rsqrtf(red[0] / (float)I_ + EPS_);
#pragma unroll
    for (int j = 0; j < 16; ++j) {
        const int i = tid + j * 256;
        ynorm[(size_t)row * I_ + i] = f32_bf16_rne(tv[j] * scale * norm_w[i]);
    }
}

// ---------------------------------------------------------------------------
// Host-side launch
// ---------------------------------------------------------------------------
extern "C" void kernel_launch(void* const* d_in, const int* in_sizes, int n_in,
                              void* d_out, int out_size, void* d_ws, size_t ws_size,
                              hipStream_t stream) {
    (void)in_sizes; (void)n_in; (void)out_size; (void)ws_size;
    const float* hs      = (const float*)d_in[0];  // (B,T,HID)
    const float* in_w    = (const float*)d_in[1];  // (TOTAL,HID)
    const float* conv_w  = (const float*)d_in[2];  // (CONV_DIM,K)
    const float* conv_b  = (const float*)d_in[3];  // (CONV_DIM,)
    const float* Av      = (const float*)d_in[4];  // (H,)
    const float* Dv      = (const float*)d_in[5];  // (H,)
    const float* dt_bias = (const float*)d_in[6];  // (H,)
    const float* norm_w  = (const float*)d_in[7];  // (I,)
    const float* out_w   = (const float*)d_in[8];  // (HID,I)
    float* out = (float*)d_out;

    // workspace carve-up (all region sizes are multiples of 256 B)
    char* ws = (char*)d_ws;
    const size_t n_hs   = (size_t)BT_ * HID_;
    const size_t n_win  = (size_t)TOTAL_ * HID_;
    const size_t n_wout = (size_t)HID_ * I_;
    unsigned short* hs_bf   = (unsigned short*)(ws);
    unsigned short* win_bf  = (unsigned short*)(ws + n_hs * 2);
    unsigned short* wout_bf = (unsigned short*)(ws + n_hs * 2 + n_win * 2);
    size_t off = n_hs * 2 + n_win * 2 + n_wout * 2;
    float* proj   = (float*)(ws + off);                 off += (size_t)BT_ * TOTAL_ * 4;
    float* convo  = (float*)(ws + off);                 off += (size_t)BT_ * CONV_DIM_ * 4;
    float* dtsp   = (float*)(ws + off);                 off += (size_t)BT_ * H_ * 4;
    float* yscan  = (float*)(ws + off);                 off += (size_t)BT_ * I_ * 4;
    unsigned short* ynorm = (unsigned short*)(ws + off);

    // 1) fp32 -> bf16
    f32_to_bf16_kernel<<<2048, 256, 0, stream>>>(hs,    hs_bf,   n_hs);
    f32_to_bf16_kernel<<<2048, 256, 0, stream>>>(in_w,  win_bf,  n_win);
    f32_to_bf16_kernel<<<2048, 256, 0, stream>>>(out_w, wout_bf, n_wout);

    // 2) in_proj: proj[BT,TOTAL] = hs_bf[BT,HID] * in_w[TOTAL,HID]^T
    {
        const int tiles = (BT_ / 64) * (TOTAL_ / 64);   // 64 * 161 = 10304
        gemm_bf16_wmma<<<(tiles + 7) / 8, 256, 0, stream>>>(
            (const bf16*)hs_bf, (const bf16*)win_bf, proj,
            BT_, TOTAL_, HID_, HID_, HID_, TOTAL_);
    }

    // 3) depthwise conv + SiLU
    {
        const size_t n = (size_t)BT_ * CONV_DIM_;
        conv_silu_kernel<<<(unsigned)((n + 255) / 256), 256, 0, stream>>>(
            proj, conv_w, conv_b, convo);
    }

    // 4) dt softplus
    {
        const size_t n = (size_t)BT_ * H_;
        dt_softplus_kernel<<<(unsigned)((n + 255) / 256), 256, 0, stream>>>(
            proj, dt_bias, dtsp);
    }

    // 5) sequential SSM scan: one block per (b,h)
    ssm_scan_kernel<<<B_ * H_, 256, 0, stream>>>(convo, dtsp, Av, Dv, yscan);

    // 6) gated RMSNorm -> bf16
    gate_norm_kernel<<<BT_, 256, 0, stream>>>(yscan, proj, norm_w, ynorm);

    // 7) out_proj: out[BT,HID] = ynorm[BT,I] * out_w[HID,I]^T
    {
        const int tiles = (BT_ / 64) * (HID_ / 64);     // 64 * 42 = 2688
        gemm_bf16_wmma<<<(tiles + 7) / 8, 256, 0, stream>>>(
            (const bf16*)ynorm, (const bf16*)wout_bf, out,
            BT_, HID_, I_, I_, I_, HID_);
    }
}
